// Qwen3AttentionFused_57535381897497
// MI455X (gfx1250) — compile-verified, matrix operand order checked by reading, not run on hardware
//
#include <hip/hip_runtime.h>
#include <hip/hip_bf16.h>

// ---------------------------------------------------------------------------
// Problem constants (from reference): T=2048, HID=2048, H=16, KV=8, D=128
// ---------------------------------------------------------------------------
constexpr int Tt   = 2048;
constexpr int HIDc = 2048;
constexpr int Hc   = 16;
constexpr int KVc  = 8;
constexpr int Dc   = 128;
constexpr int QKVN = (Hc + 2 * KVc) * Dc;   // 4096
constexpr float EPSc   = 1e-6f;
constexpr float THETAc = 10000.0f;

// ---------------------------------------------------------------------------
// WMMA types (CDNA5 gfx1250: V_WMMA_F32_16X16X32_BF16, wave32)
// ---------------------------------------------------------------------------
typedef __attribute__((ext_vector_type(16))) __bf16 bf16x16;
typedef __attribute__((ext_vector_type(8)))  float  f32x8;

union FragBF {
    bf16x16        v;
    uint4          u[2];
    unsigned short h[16];
};

__device__ inline unsigned short f2bf(float x) {
    unsigned int u = __float_as_uint(x);
    unsigned int r = (u + 0x7FFFu + ((u >> 16) & 1u)) >> 16;
    return (unsigned short)r;
}
__device__ inline float bf2f(unsigned short h) {
    return __uint_as_float(((unsigned int)h) << 16);
}

// ---------------------------------------------------------------------------
// Async global->LDS copy (GLOBAL_LOAD_ASYNC_TO_LDS_B128, ASYNCcnt-tracked).
// Builtin signature (from hipcc diagnostic): param0 = vector_size(16) int,
// address space 1 (global); param1 = same in address space 3 (LDS);
// params 2/3 = immediate offset / cache policy.
// Falls back to a synchronous VGPR round-trip if the builtins are absent.
// ---------------------------------------------------------------------------
#if __has_builtin(__builtin_amdgcn_global_load_async_to_lds_b128) && \
    __has_builtin(__builtin_amdgcn_s_wait_asynccnt)
#define USE_ASYNC_CP 1
typedef int vi4 __attribute__((vector_size(16)));
typedef __attribute__((address_space(1))) vi4* g_vi4p;
typedef __attribute__((address_space(3))) vi4* l_vi4p;
#else
#define USE_ASYNC_CP 0
#endif

__device__ inline void cp_g2l_16(void* lds, const void* g) {
#if USE_ASYNC_CP
    __builtin_amdgcn_global_load_async_to_lds_b128(
        (g_vi4p)g, (l_vi4p)lds, 0, 0);
#else
    *(uint4*)lds = *(const uint4*)g;
#endif
}
__device__ inline void cp_g2l_join() {
#if USE_ASYNC_CP
    __builtin_amdgcn_s_wait_asynccnt(0);   // s_wait_asynccnt 0
#endif
}

// A/B fragment load from an LDS tile (row-major, `stride` halves per row).
// ISA 7.12.2: lanes 0-15 hold row/col = lane, K in {0..7, 16..23};
//             lanes 16-31 hold row/col = lane-16, K in {8..15, 24..31}.
__device__ inline void load_frag(const unsigned short* tile, int row0, int k0,
                                 int stride, int lane, FragBF& f) {
    int r  = row0 + (lane & 15);
    int kh = k0 + ((lane & 16) ? 8 : 0);
    const unsigned short* p = tile + (size_t)r * stride + kh;
    f.u[0] = *(const uint4*)(p);
    f.u[1] = *(const uint4*)(p + 16);
}

__device__ inline float hmax16(float v) {
    #pragma unroll
    for (int m = 1; m < 16; m <<= 1) v = fmaxf(v, __shfl_xor(v, m, 32));
    return v;
}
__device__ inline float hsum16(float v) {
    #pragma unroll
    for (int m = 1; m < 16; m <<= 1) v += __shfl_xor(v, m, 32);
    return v;
}

// ---------------------------------------------------------------------------
// Kernel 1: row RMSNorm of hidden_states -> bf16
// ---------------------------------------------------------------------------
__global__ void rmsnorm_bf16(const float* __restrict__ x,
                             const float* __restrict__ w,
                             unsigned short* __restrict__ y) {
    int t = blockIdx.x;
    const float* row = x + (size_t)t * HIDc;
    float ss = 0.0f;
    for (int i = threadIdx.x; i < HIDc; i += 256) {
        float v = row[i];
        ss += v * v;
    }
    #pragma unroll
    for (int m = 1; m < 32; m <<= 1) ss += __shfl_xor(ss, m, 32);
    __shared__ float red[8];
    __shared__ float stotal;
    int lane = threadIdx.x & 31, warp = threadIdx.x >> 5;
    if (lane == 0) red[warp] = ss;
    __syncthreads();
    if (threadIdx.x == 0) {
        float s = 0.0f;
        #pragma unroll
        for (int i = 0; i < 8; ++i) s += red[i];
        stotal = s;
    }
    __syncthreads();
    float inv = rsqrtf(stotal / (float)HIDc + EPSc);
    for (int i = threadIdx.x; i < HIDc; i += 256)
        y[(size_t)t * HIDc + i] = f2bf(row[i] * inv * w[i]);
}

// ---------------------------------------------------------------------------
// Kernel 2: fp32 -> bf16 conversion (weights)
// ---------------------------------------------------------------------------
__global__ void f32_to_bf16(const float* __restrict__ x,
                            unsigned short* __restrict__ y, int n) {
    int i = blockIdx.x * 256 + threadIdx.x;
    if (i < n) y[i] = f2bf(x[i]);
}

// ---------------------------------------------------------------------------
// Kernel 3/6: bf16 GEMM  C[M,N] = A[M,K] * B[N,K]^T  via v_wmma_f32_16x16x32_bf16
// Block: 256 threads (8 waves), tile 128x128, K-step 32. Waves: 4 (M) x 2 (N),
// each wave owns a 32x64 sub-tile = 2x4 WMMA accumulators.
// ---------------------------------------------------------------------------
constexpr int GBM = 128, GBN = 128, GBK = 32, GLS = 40; // 40-half padded stride

template <typename OUT>
__device__ inline void store_c(OUT* C, size_t idx, float v);
template <> __device__ inline void store_c<float>(float* C, size_t idx, float v) { C[idx] = v; }
template <> __device__ inline void store_c<unsigned short>(unsigned short* C, size_t idx, float v) { C[idx] = f2bf(v); }

template <typename OUT>
__global__ void __launch_bounds__(256, 2)
gemm_bf16(const unsigned short* __restrict__ A,
          const unsigned short* __restrict__ B,
          OUT* __restrict__ C,
          int M, int N, int K) {
    __shared__ unsigned short As[GBM * GLS];
    __shared__ unsigned short Bs[GBN * GLS];

    int tid  = threadIdx.x;
    int lane = tid & 31;
    int wave = tid >> 5;
    int wm   = wave >> 1;        // 0..3 -> 32-row slab
    int wn   = wave & 1;         // 0..1 -> 64-col slab
    int bm   = blockIdx.y * GBM;
    int bn   = blockIdx.x * GBN;

    f32x8 zero = {0, 0, 0, 0, 0, 0, 0, 0};
    f32x8 acc[2][4];
    #pragma unroll
    for (int mi = 0; mi < 2; ++mi)
        #pragma unroll
        for (int nj = 0; nj < 4; ++nj) acc[mi][nj] = zero;

    for (int kb = 0; kb < K; kb += GBK) {
        __syncthreads();
        // Stage A & B tiles: 128 rows x 32 halves = 512 x 16B chunks each,
        // via async global->LDS copies (ASYNCcnt) when available.
        #pragma unroll
        for (int i = 0; i < 2; ++i) {
            int c  = tid + i * 256;
            int r  = c >> 2;
            int kc = (c & 3) * 8;
            cp_g2l_16(&As[(size_t)r * GLS + kc],
                      A + (size_t)(bm + r) * K + kb + kc);
            cp_g2l_16(&Bs[(size_t)r * GLS + kc],
                      B + (size_t)(bn + r) * K + kb + kc);
        }
        cp_g2l_join();
        __syncthreads();

        FragBF af[2], bf[4];
        #pragma unroll
        for (int mi = 0; mi < 2; ++mi)
            load_frag(As, wm * 32 + mi * 16, 0, GLS, lane, af[mi]);
        #pragma unroll
        for (int nj = 0; nj < 4; ++nj)
            load_frag(Bs, wn * 64 + nj * 16, 0, GLS, lane, bf[nj]);

        #pragma unroll
        for (int mi = 0; mi < 2; ++mi)
            #pragma unroll
            for (int nj = 0; nj < 4; ++nj)
                acc[mi][nj] = __builtin_amdgcn_wmma_f32_16x16x32_bf16(
                    false, af[mi].v, false, bf[nj].v,
                    (short)0, acc[mi][nj], false, false);
    }

    // C layout: VGPR i -> M = i (+8 for upper lane half), lane&15 -> N
    int rhalf = (lane & 16) ? 8 : 0;
    #pragma unroll
    for (int mi = 0; mi < 2; ++mi)
        #pragma unroll
        for (int nj = 0; nj < 4; ++nj)
            #pragma unroll
            for (int i = 0; i < 8; ++i) {
                int r = bm + wm * 32 + mi * 16 + rhalf + i;
                int c = bn + wn * 64 + nj * 16 + (lane & 15);
                store_c<OUT>(C, (size_t)r * N + c, acc[mi][nj][i]);
            }
}

// ---------------------------------------------------------------------------
// Kernel 4: per-(token, head) RMSNorm of q/k heads + RoPE  -> bf16 q/k tensors
// grid (T, H+KV), block 128 (one thread per channel d)
// ---------------------------------------------------------------------------
__global__ void qknorm_rope(const unsigned short* __restrict__ qkv,
                            const int* __restrict__ positions,
                            const float* __restrict__ qw,
                            const float* __restrict__ kw,
                            unsigned short* __restrict__ qout,
                            unsigned short* __restrict__ kout) {
    int t  = blockIdx.x;
    int hh = blockIdx.y;
    int d  = threadIdx.x;
    bool isq = hh < Hc;
    int off  = isq ? hh * Dc : Hc * Dc + (hh - Hc) * Dc;

    float v  = bf2f(qkv[(size_t)t * QKVN + off + d]);
    float ss = v * v;
    #pragma unroll
    for (int m = 1; m < 32; m <<= 1) ss += __shfl_xor(ss, m, 32);
    __shared__ float red[4];
    __shared__ float stotal;
    if ((d & 31) == 0) red[d >> 5] = ss;
    __syncthreads();
    if (d == 0) stotal = red[0] + red[1] + red[2] + red[3];
    __syncthreads();
    float inv = rsqrtf(stotal / (float)Dc + EPSc);
    const float* w = isq ? qw : kw;

    __shared__ float buf[Dc];
    buf[d] = v * inv * w[d];
    __syncthreads();

    int   j    = d & 63;
    float freq = __powf(THETAc, -((float)(2 * j)) / (float)Dc);
    float ang  = (float)positions[t] * freq;
    float c, s;
    __sincosf(ang, &s, &c);
    float x1 = buf[j], x2 = buf[j + 64];
    float o  = (d < 64) ? (x1 * c - x2 * s) : (x2 * c + x1 * s);

    unsigned short ob = f2bf(o);
    if (isq) qout[((size_t)t * Hc + hh) * Dc + d] = ob;
    else     kout[((size_t)t * KVc + (hh - Hc)) * Dc + d] = ob;
}

// ---------------------------------------------------------------------------
// Kernel 5: causal flash attention, one wave (32 threads) per (16-q tile, head)
// S = Q K^T (WMMA), online softmax (16-lane shfl reductions), O += P V (WMMA)
// __launch_bounds__(32,1): single wave per SIMD -> full VGPR budget, no spills
// ---------------------------------------------------------------------------
constexpr int KST = 136;  // K/V LDS tile stride (halves), 16B-aligned rows
constexpr int PST = 40;   // P LDS tile stride

__global__ void __launch_bounds__(32, 1)
flash_attn(const unsigned short* __restrict__ qb,
           const unsigned short* __restrict__ kb_,
           const unsigned short* __restrict__ qkv,
           unsigned short* __restrict__ attn) {
    __shared__ unsigned short Qs[16 * Dc];
    __shared__ unsigned short Ks[32 * KST];
    __shared__ unsigned short Vs[32 * KST];
    __shared__ unsigned short Ps[16 * PST];

    int lane = threadIdx.x & 31;
    int qt   = blockIdx.x;       // 16-row query tile
    int h    = blockIdx.y;       // query head
    int kvh  = h / (Hc / KVc);   // kv head (GQA repeat)
    int qlast = qt * 16 + 15;

    // Stage Q tile (16 x 128 halves = 256 x 16B chunks)
    for (int c = lane; c < 256; c += 32) {
        int r = c >> 4, kc = (c & 15) * 8;
        cp_g2l_16(&Qs[(size_t)r * Dc + kc],
                  qb + ((size_t)(qt * 16 + r) * Hc + h) * Dc + kc);
    }
    cp_g2l_join();
    __syncthreads();

    FragBF qf[4];
    #pragma unroll
    for (int ks = 0; ks < 4; ++ks) load_frag(Qs, 0, ks * 32, Dc, lane, qf[ks]);

    f32x8 zero = {0, 0, 0, 0, 0, 0, 0, 0};
    f32x8 Oacc[8];
    #pragma unroll
    for (int nj = 0; nj < 8; ++nj) Oacc[nj] = zero;
    float mrow[8], lrow[8];
    #pragma unroll
    for (int i = 0; i < 8; ++i) { mrow[i] = -3.0e38f; lrow[i] = 0.0f; }

    int rhalf = (lane & 16) ? 8 : 0;

    for (int kb = 0; kb <= qlast; kb += 32) {
        __syncthreads();
        // Stage K and V tiles: 32 rows x 128 halves = 512 x 16B chunks each.
        for (int c = lane; c < 512; c += 32) {
            int r = c >> 4, kc = (c & 15) * 8;
            int t = kb + r;   // always < T by causal loop bound
            cp_g2l_16(&Ks[(size_t)r * KST + kc],
                      kb_ + ((size_t)t * KVc + kvh) * Dc + kc);
            cp_g2l_16(&Vs[(size_t)r * KST + kc],
                      qkv + (size_t)t * QKVN + (Hc + KVc) * Dc + kvh * Dc + kc);
        }
        cp_g2l_join();
        __syncthreads();

        // S tiles for the two 16-key halves
        f32x8 s0 = zero, s1 = zero;
        #pragma unroll
        for (int ks = 0; ks < 4; ++ks) {
            FragBF kf0, kf1;
            load_frag(Ks, 0,  ks * 32, KST, lane, kf0);
            load_frag(Ks, 16, ks * 32, KST, lane, kf1);
            s0 = __builtin_amdgcn_wmma_f32_16x16x32_bf16(false, qf[ks].v, false, kf0.v, (short)0, s0, false, false);
            s1 = __builtin_amdgcn_wmma_f32_16x16x32_bf16(false, qf[ks].v, false, kf1.v, (short)0, s1, false, false);
        }

        const float scale = 0.08838834764831845f; // 1/sqrt(128)
        int key0 = kb + (lane & 15);
        int key1 = key0 + 16;
        #pragma unroll
        for (int i = 0; i < 8; ++i) {
            int r = qt * 16 + rhalf + i;
            float v0 = (key0 <= r) ? s0[i] * scale : -3.0e38f;
            float v1 = (key1 <= r) ? s1[i] * scale : -3.0e38f;
            float rm   = hmax16(fmaxf(v0, v1));
            float mnew = fmaxf(mrow[i], rm);
            float sc   = __expf(mrow[i] - mnew);
            float p0   = __expf(v0 - mnew);
            float p1   = __expf(v1 - mnew);
            lrow[i] = lrow[i] * sc + hsum16(p0) + hsum16(p1);
            mrow[i] = mnew;
            #pragma unroll
            for (int nj = 0; nj < 8; ++nj) Oacc[nj][i] *= sc;
            Ps[(size_t)(rhalf + i) * PST + (lane & 15)]      = f2bf(p0);
            Ps[(size_t)(rhalf + i) * PST + (lane & 15) + 16] = f2bf(p1);
        }

        // O += P * V : P as A-fragment (16x32 keys), V gathered as B-fragments
        FragBF pf;
        load_frag(Ps, 0, 0, PST, lane, pf);
        int kbase = (lane & 16) ? 8 : 0;
        #pragma unroll
        for (int nj = 0; nj < 8; ++nj) {
            FragBF vf;
            int d = nj * 16 + (lane & 15);
            #pragma unroll
            for (int j = 0; j < 16; ++j) {
                int key = kbase + ((j < 8) ? j : j + 8);
                vf.h[j] = Vs[(size_t)key * KST + d];
            }
            Oacc[nj] = __builtin_amdgcn_wmma_f32_16x16x32_bf16(
                false, pf.v, false, vf.v, (short)0, Oacc[nj], false, false);
        }
    }

    // Normalize and write attn output (bf16, [T, H*D]) for the O-projection
    #pragma unroll
    for (int i = 0; i < 8; ++i) {
        float inv = 1.0f / lrow[i];
        int t = qt * 16 + rhalf + i;
        #pragma unroll
        for (int nj = 0; nj < 8; ++nj) {
            int c = h * Dc + nj * 16 + (lane & 15);
            attn[(size_t)t * (Hc * Dc) + c] = f2bf(Oacc[nj][i] * inv);
        }
    }
}

// ---------------------------------------------------------------------------
// Host launcher
// ---------------------------------------------------------------------------
extern "C" void kernel_launch(void* const* d_in, const int* in_sizes, int n_in,
                              void* d_out, int out_size, void* d_ws, size_t ws_size,
                              hipStream_t stream) {
    const int*   positions = (const int*)d_in[0];
    const float* hidden    = (const float*)d_in[1];
    const float* lnw       = (const float*)d_in[2];
    const float* qkvw_f    = (const float*)d_in[3];
    const float* qnw       = (const float*)d_in[4];
    const float* knw       = (const float*)d_in[5];
    const float* ow_f      = (const float*)d_in[6];
    float*       out       = (float*)d_out;

    unsigned short* ws = (unsigned short*)d_ws;
    size_t off = 0;
    auto alloc = [&](size_t n) { unsigned short* p = ws + off; off += n; return p; };
    unsigned short* normed = alloc((size_t)Tt * HIDc);       //  8 MB
    unsigned short* qkvw_b = alloc((size_t)QKVN * HIDc);     // 16 MB
    unsigned short* ow_b   = alloc((size_t)HIDc * Hc * Dc);  //  8 MB
    unsigned short* qkv_b  = alloc((size_t)Tt * QKVN);       // 16 MB
    unsigned short* q_b    = alloc((size_t)Tt * Hc * Dc);    //  8 MB
    unsigned short* k_b    = alloc((size_t)Tt * KVc * Dc);   //  4 MB
    unsigned short* attn_b = alloc((size_t)Tt * Hc * Dc);    //  8 MB
    (void)ws_size; (void)in_sizes; (void)n_in; (void)out_size;

    // 1. RMSNorm -> bf16
    rmsnorm_bf16<<<Tt, 256, 0, stream>>>(hidden, lnw, normed);

    // 2. Weight conversions
    {
        int n = QKVN * HIDc;
        f32_to_bf16<<<(n + 255) / 256, 256, 0, stream>>>(qkvw_f, qkvw_b, n);
        n = HIDc * Hc * Dc;
        f32_to_bf16<<<(n + 255) / 256, 256, 0, stream>>>(ow_f, ow_b, n);
    }

    // 3. QKV projection: [T,HID] x [QKVN,HID]^T -> [T,QKVN] (bf16 out)
    gemm_bf16<unsigned short><<<dim3(QKVN / GBN, Tt / GBM), 256, 0, stream>>>(
        normed, qkvw_b, qkv_b, Tt, QKVN, HIDc);

    // 4. q/k head RMSNorm + RoPE
    qknorm_rope<<<dim3(Tt, Hc + KVc), 128, 0, stream>>>(
        qkv_b, positions, qnw, knw, q_b, k_b);

    // 5. Causal flash attention
    flash_attn<<<dim3(Tt / 16, Hc), 32, 0, stream>>>(q_b, k_b, qkv_b, attn_b);

    // 6. Output projection: [T,H*D] x [HID,H*D]^T -> [T,HID] (fp32 out)
    gemm_bf16<float><<<dim3(HIDc / GBN, Tt / GBM), 256, 0, stream>>>(
        attn_b, ow_b, out, Tt, HIDc, Hc * Dc);
}
